// EGNN_cont_8366596292980
// MI455X (gfx1250) — matrix operand order repaired and checked by reading, not compiled, not run on hardware
//
#include <hip/hip_runtime.h>

typedef __attribute__((ext_vector_type(16))) _Float16 v16h;
typedef __attribute__((ext_vector_type(4)))  _Float16 v4h;
typedef __attribute__((ext_vector_type(8)))  float    v8f;

// ---------------------------------------------------------------------------
// WMMA fragment helpers (CDNA5 v_wmma_f32_16x16x32_f16, wave32; A/B use this code's contiguous K order)
// A (16x32 f16): lane L -> row L&15, 16 contiguous K starting at 16*(L>>4).
// B (32x16 f16): lane L -> col L&15, 16 contiguous K starting at 16*(L>>4).
// C/D (16x16 f32): elem i of lane L -> (m = 8*(L>>4)+i, n = L&15).
// ---------------------------------------------------------------------------
__device__ __forceinline__ v16h lda(const _Float16* base, int stride, int k0) {
  const int lane = threadIdx.x & 31;
  const _Float16* p = base + (lane & 15) * stride + k0 + ((lane >> 4) << 4);
  v16h a;
#pragma unroll
  for (int i = 0; i < 8; ++i) a[i] = p[i];
#pragma unroll
  for (int i = 0; i < 8; ++i) a[i + 8] = p[i + 8];
  return a;
}

__device__ __forceinline__ v16h ldb(const _Float16* __restrict__ wt, int kpad,
                                    int n0, int k0) {
  const int lane = threadIdx.x & 31;
  const _Float16* p = wt + (size_t)(n0 + (lane & 15)) * kpad + k0 + ((lane >> 4) << 4);
  v16h b;
#pragma unroll
  for (int i = 0; i < 16; ++i) b[i] = p[i];
  return b;
}

__device__ __forceinline__ v8f wmma32(v16h a, v16h b, v8f c) {
  return __builtin_amdgcn_wmma_f32_16x16x32_f16(false, a, false, b, (short)0, c,
                                                false, false);
}

__device__ __forceinline__ void st_act(v8f d, const float* __restrict__ bias,
                                       int n0, _Float16* out, int stride, bool act) {
  const int lane = threadIdx.x & 31;
  const int n = lane & 15, mb = (lane >> 4) << 3;
  const float bv = bias[n0 + n];
#pragma unroll
  for (int i = 0; i < 8; ++i) {
    float v = d[i] + bv;
    if (act) v = v > 0.f ? v : 0.3f * v;
    out[(mb + i) * stride + n0 + n] = (_Float16)v;
  }
}

__device__ __forceinline__ v4h cvt4(float4 v) {
  v4h r;
  r[0] = (_Float16)v.x; r[1] = (_Float16)v.y;
  r[2] = (_Float16)v.z; r[3] = (_Float16)v.w;
  return r;
}

// ---------------------------------------------------------------------------
// Weight prep: f32 row-major [K][N] -> f16 transposed+padded [Npad][Kpad],
// with input-feature permutation folded into the K index:
//  mode 0: identity
//  mode 1 (eW1): src order [t, h_r(64), t, h_c(64), radial, ea(4)]
//                dst order [h_r(64) | h_c(64) | t, t, radial, ea(4) | pad]
//  mode 2 (nW1): src order [t, h(64), aggs(16)]
//                dst order [h(64) | aggs(16) | t | pad]
// ---------------------------------------------------------------------------
__global__ void prep_w(const float* __restrict__ src, _Float16* __restrict__ dst,
                       int K, int Nn, int Kp, int Np, int mode) {
  int i = blockIdx.x * blockDim.x + threadIdx.x;
  if (i >= Np * Kp) return;
  int n = i / Kp, k = i - n * Kp;
  int ks;
  if (mode == 1) {
    if (k < 128)       ks = (k < 64) ? k + 1 : k + 2;  // h_r, h_c
    else if (k == 128) ks = 0;                          // t
    else if (k == 129) ks = 65;                         // t (second copy)
    else if (k <= 134) ks = k;                          // radial(130), ea(131..134)
    else               ks = -1;
  } else if (mode == 2) {
    if (k < 80)        ks = k + 1;                      // h(64) + aggs(16)
    else if (k == 80)  ks = 0;                          // t
    else               ks = -1;
  } else {
    ks = (k < K) ? k : -1;
  }
  float v = (ks >= 0 && ks < K && n < Nn) ? src[(size_t)ks * Nn + n] : 0.f;
  dst[i] = (_Float16)v;
}

// ---------------------------------------------------------------------------
// Embedding: h0 = x @ emb_W + emb_b ; state init ; traj[0] = h0
// ---------------------------------------------------------------------------
__global__ void embed_kernel(const float* __restrict__ x,
                             const float* __restrict__ emb_W,
                             const float* __restrict__ emb_b,
                             const float* __restrict__ pos,
                             float* __restrict__ H0, float* __restrict__ P0,
                             float* __restrict__ traj, int N) {
  int idx = blockIdx.x * blockDim.x + threadIdx.x;
  if (idx < N * 64) {
    int node = idx >> 6, f = idx & 63;
    float acc = emb_b[f];
    const float* xr = x + (size_t)node * 128;
#pragma unroll 8
    for (int k = 0; k < 128; ++k) acc = fmaf(xr[k], emb_W[k * 64 + f], acc);
    H0[idx] = acc;
    traj[idx] = acc;
  }
  if (idx < N * 3) P0[idx] = pos[idx];
}

__global__ void count_kernel(const int* __restrict__ eidx, float* __restrict__ cntf,
                             int E) {
  int e = blockIdx.x * blockDim.x + threadIdx.x;
  if (e < E) atomicAdd(&cntf[eidx[e]], 1.0f);
}

// ---------------------------------------------------------------------------
// Edge kernel: one wave32 per 16-edge tile. Fused:
//   ein (permuted, 135->160) -> eMLP 160->64->32->16 -> cMLP 32->32->32->3(16)
//   atomics: ef -> aggs[row], clip(cdiff*cmlp) -> csum[row]
// ---------------------------------------------------------------------------
__global__ void __launch_bounds__(32)
edge_kernel(const float* __restrict__ H, const float* __restrict__ P,
            const float* __restrict__ ea, const int* __restrict__ eidx,
            const _Float16* __restrict__ eW1t, const float* __restrict__ eb1,
            const _Float16* __restrict__ eW2t, const float* __restrict__ eb2,
            const _Float16* __restrict__ eW3t, const float* __restrict__ eb3,
            const _Float16* __restrict__ cW1t, const float* __restrict__ cb1,
            const _Float16* __restrict__ cW2t, const float* __restrict__ cb2,
            const _Float16* __restrict__ cW3t, const float* __restrict__ cb3,
            float* __restrict__ aggs, float* __restrict__ csum,
            float t, int E) {
  __shared__ _Float16 sIn[16][160];
  __shared__ _Float16 sA[16][64];
  __shared__ _Float16 sB[16][64];
  __shared__ float sCd[16][3];
  __shared__ int sRow[16];
  __shared__ int sColN[16];

  const int lane = threadIdx.x;
  const int e0 = blockIdx.x * 16;

  // phase 0: per-edge scalars; lanes 0..15 stash indices/cdiff,
  // lanes 16..31 fill the tail columns [t, t, radial, ea, zeros]
  {
    const int m = lane & 15;
    const int e = min(e0 + m, E - 1);
    const int r = eidx[e];
    const int c = eidx[E + e];
    const float d0 = P[r * 3 + 0] - P[c * 3 + 0];
    const float d1 = P[r * 3 + 1] - P[c * 3 + 1];
    const float d2 = P[r * 3 + 2] - P[c * 3 + 2];
    const float radial = d0 * d0 + d1 * d1 + d2 * d2;
    if (lane < 16) {
      sRow[m] = r;
      sColN[m] = c;
      sCd[m][0] = d0; sCd[m][1] = d1; sCd[m][2] = d2;
    } else {
      float4 av = *(const float4*)(ea + (size_t)e * 4);
      _Float16* tail = &sIn[m][128];
      tail[0] = (_Float16)t;
      tail[1] = (_Float16)t;
      tail[2] = (_Float16)radial;
      tail[3] = (_Float16)av.x;
      tail[4] = (_Float16)av.y;
      tail[5] = (_Float16)av.z;
      tail[6] = (_Float16)av.w;
#pragma unroll
      for (int j = 7; j < 32; ++j) tail[j] = (_Float16)0.f;
    }
  }
  __syncthreads();

  // phase 1: coalesced h gathers. Each half-wave streams one node row
  // (64 floats = 256B) via float4, packed to f16 in LDS via v4h stores.
#pragma unroll
  for (int it = 0; it < 8; ++it) {
    const int id = it * 32 + lane;
    const int m = id >> 4;
    const int c4 = (id & 15) << 2;
    float4 vr = *(const float4*)(H + (size_t)sRow[m] * 64 + c4);
    float4 vc = *(const float4*)(H + (size_t)sColN[m] * 64 + c4);
    *(v4h*)&sIn[m][c4] = cvt4(vr);
    *(v4h*)&sIn[m][64 + c4] = cvt4(vc);
  }
  __syncthreads();

  // edge L1: 160 -> 64 (lrelu)
  v16h af[5];
#pragma unroll
  for (int kt = 0; kt < 5; ++kt) af[kt] = lda(&sIn[0][0], 160, kt * 32);
#pragma unroll
  for (int nt = 0; nt < 4; ++nt) {
    v8f acc = {};
#pragma unroll
    for (int kt = 0; kt < 5; ++kt)
      acc = wmma32(af[kt], ldb(eW1t, 160, nt * 16, kt * 32), acc);
    st_act(acc, eb1, nt * 16, &sA[0][0], 64, true);
  }
  __syncthreads();

  // edge L2: 64 -> 32 (lrelu)
  {
    v16h a0 = lda(&sA[0][0], 64, 0);
    v16h a1 = lda(&sA[0][0], 64, 32);
#pragma unroll
    for (int nt = 0; nt < 2; ++nt) {
      v8f acc = {};
      acc = wmma32(a0, ldb(eW2t, 64, nt * 16, 0), acc);
      acc = wmma32(a1, ldb(eW2t, 64, nt * 16, 32), acc);
      st_act(acc, eb2, nt * 16, &sB[0][0], 64, true);
    }
  }
  __syncthreads();

  const int n = lane & 15, mb = (lane >> 4) << 3;

  // edge L3: 32 -> 16 (ef, no activation) ; scatter ef -> aggs
  {
    v16h a = lda(&sB[0][0], 64, 0);
    v8f acc = {};
    acc = wmma32(a, ldb(eW3t, 32, 0, 0), acc);
    const float bv = eb3[n];
#pragma unroll
    for (int i = 0; i < 8; ++i) {
      float v = acc[i] + bv;
      sA[mb + i][n] = (_Float16)v;        // ef as f16, K-padded to 32
      sA[mb + i][16 + n] = (_Float16)0.f;
      if (e0 + mb + i < E)
        atomicAdd(&aggs[(size_t)sRow[mb + i] * 16 + n], v);
    }
  }
  __syncthreads();

  // coord c1: 16(pad32) -> 32 (lrelu)
  {
    v16h a = lda(&sA[0][0], 64, 0);
#pragma unroll
    for (int nt = 0; nt < 2; ++nt) {
      v8f acc = {};
      acc = wmma32(a, ldb(cW1t, 32, nt * 16, 0), acc);
      st_act(acc, cb1, nt * 16, &sB[0][0], 64, true);
    }
  }
  __syncthreads();

  // coord c2: 32 -> 32 (lrelu)
  {
    v16h a = lda(&sB[0][0], 64, 0);
#pragma unroll
    for (int nt = 0; nt < 2; ++nt) {
      v8f acc = {};
      acc = wmma32(a, ldb(cW2t, 32, nt * 16, 0), acc);
      st_act(acc, cb2, nt * 16, &sA[0][0], 64, true);
    }
  }
  __syncthreads();

  // coord c3: 32 -> 3 (N padded to 16) ; trans = clip(cdiff*out) -> csum
  {
    v16h a = lda(&sA[0][0], 64, 0);
    v8f acc = {};
    acc = wmma32(a, ldb(cW3t, 32, 0, 0), acc);
    if (n < 3) {
      const float bv = cb3[n];
#pragma unroll
      for (int i = 0; i < 8; ++i) {
        const int m = mb + i;
        if (e0 + m < E) {
          float tr = sCd[m][n] * (acc[i] + bv);
          tr = fminf(fmaxf(tr, -100.f), 100.f);
          atomicAdd(&csum[(size_t)sRow[m] * 3 + n], tr);
        }
      }
    }
  }
}

// ---------------------------------------------------------------------------
// Node kernel: one wave32 per 16-node tile.
//   nin (permuted, 81->96) -> nMLP 96->128->128->64
//   h' = h + dt*nh ; p' = p + dt*(p + csum/cnt)
// ---------------------------------------------------------------------------
__global__ void __launch_bounds__(32)
node_kernel(const float* __restrict__ Hin, const float* __restrict__ Pin,
            const float* __restrict__ aggs, const float* __restrict__ csum,
            const float* __restrict__ cntf,
            const _Float16* __restrict__ nW1t, const float* __restrict__ nb1,
            const _Float16* __restrict__ nW2t, const float* __restrict__ nb2,
            const _Float16* __restrict__ nW3t, const float* __restrict__ nb3,
            float* __restrict__ Hout, float* __restrict__ Pout,
            float* __restrict__ traj, float t, float dt, int N) {
  __shared__ _Float16 sIn[16][96];
  __shared__ _Float16 sA[16][128];
  __shared__ _Float16 sB[16][128];

  const int lane = threadIdx.x;
  const int b0 = blockIdx.x * 16;

  // tail columns [t | zeros]
  if (lane >= 16) {
    _Float16* tail = &sIn[lane & 15][80];
    tail[0] = (_Float16)t;
#pragma unroll
    for (int j = 1; j < 16; ++j) tail[j] = (_Float16)0.f;
  }
  // coalesced h gather: cols 0..63
#pragma unroll
  for (int it = 0; it < 8; ++it) {
    const int id = it * 32 + lane;
    const int m = id >> 4;
    const int c4 = (id & 15) << 2;
    const int node = min(b0 + m, N - 1);
    float4 v = *(const float4*)(Hin + (size_t)node * 64 + c4);
    *(v4h*)&sIn[m][c4] = cvt4(v);
  }
  // aggs gather: cols 64..79
#pragma unroll
  for (int it = 0; it < 2; ++it) {
    const int id = it * 32 + lane;
    const int m = id >> 2;
    const int c4 = (id & 3) << 2;
    const int node = min(b0 + m, N - 1);
    float4 v = *(const float4*)(aggs + (size_t)node * 16 + c4);
    *(v4h*)&sIn[m][64 + c4] = cvt4(v);
  }
  __syncthreads();

  // node L1: 96 -> 128 (lrelu)
  {
    v16h af[3];
#pragma unroll
    for (int kt = 0; kt < 3; ++kt) af[kt] = lda(&sIn[0][0], 96, kt * 32);
#pragma unroll
    for (int nt = 0; nt < 8; ++nt) {
      v8f acc = {};
#pragma unroll
      for (int kt = 0; kt < 3; ++kt)
        acc = wmma32(af[kt], ldb(nW1t, 96, nt * 16, kt * 32), acc);
      st_act(acc, nb1, nt * 16, &sA[0][0], 128, true);
    }
  }
  __syncthreads();

  // node L2: 128 -> 128 (lrelu)
  {
    v16h af[4];
#pragma unroll
    for (int kt = 0; kt < 4; ++kt) af[kt] = lda(&sA[0][0], 128, kt * 32);
#pragma unroll
    for (int nt = 0; nt < 8; ++nt) {
      v8f acc = {};
#pragma unroll
      for (int kt = 0; kt < 4; ++kt)
        acc = wmma32(af[kt], ldb(nW2t, 128, nt * 16, kt * 32), acc);
      st_act(acc, nb2, nt * 16, &sB[0][0], 128, true);
    }
  }
  __syncthreads();

  // node L3: 128 -> 64 ; Euler update + traj write
  {
    const int n = lane & 15, mb = (lane >> 4) << 3;
    v16h af[4];
#pragma unroll
    for (int kt = 0; kt < 4; ++kt) af[kt] = lda(&sB[0][0], 128, kt * 32);
#pragma unroll
    for (int nt = 0; nt < 4; ++nt) {
      v8f acc = {};
#pragma unroll
      for (int kt = 0; kt < 4; ++kt)
        acc = wmma32(af[kt], ldb(nW3t, 128, nt * 16, kt * 32), acc);
      const int f = nt * 16 + n;
      const float bv = nb3[f];
#pragma unroll
      for (int i = 0; i < 8; ++i) {
        const int node = b0 + mb + i;
        if (node < N) {
          float hnew = Hin[(size_t)node * 64 + f] + dt * (acc[i] + bv);
          Hout[(size_t)node * 64 + f] = hnew;
          traj[(size_t)node * 64 + f] = hnew;
        }
      }
    }
  }

  // coords: p' = p + dt * (p + csum/cnt)
  for (int q = lane; q < 48; q += 32) {
    const int node = b0 + q / 3;
    const int d = q % 3;
    if (node < N) {
      float p = Pin[(size_t)node * 3 + d];
      float cnt = fmaxf(cntf[node], 1.0f);
      float coord = p + csum[(size_t)node * 3 + d] / cnt;
      Pout[(size_t)node * 3 + d] = p + dt * coord;
    }
  }
}

// ---------------------------------------------------------------------------
extern "C" void kernel_launch(void* const* d_in, const int* in_sizes, int n_in,
                              void* d_out, int out_size, void* d_ws, size_t ws_size,
                              hipStream_t stream) {
  (void)in_sizes; (void)n_in; (void)out_size; (void)ws_size;
  const int N = 50000, E = 800000;

  const float* x     = (const float*)d_in[0];
  const float* pos   = (const float*)d_in[1];
  const float* ea    = (const float*)d_in[2];
  const int*   eidx  = (const int*)d_in[3];
  const float* emb_W = (const float*)d_in[4];
  const float* emb_b = (const float*)d_in[5];
  const float* nW1 = (const float*)d_in[6];  const float* nb1 = (const float*)d_in[7];
  const float* nW2 = (const float*)d_in[8];  const float* nb2 = (const float*)d_in[9];
  const float* nW3 = (const float*)d_in[10]; const float* nb3 = (const float*)d_in[11];
  const float* eW1 = (const float*)d_in[12]; const float* eb1 = (const float*)d_in[13];
  const float* eW2 = (const float*)d_in[14]; const float* eb2 = (const float*)d_in[15];
  const float* eW3 = (const float*)d_in[16]; const float* eb3 = (const float*)d_in[17];
  const float* cW1 = (const float*)d_in[18]; const float* cb1 = (const float*)d_in[19];
  const float* cW2 = (const float*)d_in[20]; const float* cb2 = (const float*)d_in[21];
  const float* cW3 = (const float*)d_in[22]; const float* cb3 = (const float*)d_in[23];

  char* ws = (char*)d_ws;
  size_t off = 0;
  auto carve = [&](size_t bytes) {
    char* p = ws + off;
    off = (off + bytes + 255) & ~(size_t)255;
    return p;
  };
  float* H0   = (float*)carve((size_t)N * 64 * 4);
  float* H1   = (float*)carve((size_t)N * 64 * 4);
  float* P0   = (float*)carve((size_t)N * 3 * 4);
  float* P1   = (float*)carve((size_t)N * 3 * 4);
  float* aggs = (float*)carve((size_t)N * 16 * 4);
  float* csum = (float*)carve((size_t)N * 3 * 4);
  float* cntf = (float*)carve((size_t)N * 4);
  _Float16* eW1t = (_Float16*)carve(64 * 160 * 2);
  _Float16* eW2t = (_Float16*)carve(32 * 64 * 2);
  _Float16* eW3t = (_Float16*)carve(16 * 32 * 2);
  _Float16* cW1t = (_Float16*)carve(32 * 32 * 2);
  _Float16* cW2t = (_Float16*)carve(32 * 32 * 2);
  _Float16* cW3t = (_Float16*)carve(16 * 32 * 2);
  _Float16* nW1t = (_Float16*)carve(128 * 96 * 2);
  _Float16* nW2t = (_Float16*)carve(128 * 128 * 2);
  _Float16* nW3t = (_Float16*)carve(64 * 128 * 2);

  auto prep = [&](const float* src, _Float16* dst, int K, int Nn, int Kp, int Np,
                  int mode) {
    int tot = Np * Kp;
    prep_w<<<(tot + 255) / 256, 256, 0, stream>>>(src, dst, K, Nn, Kp, Np, mode);
  };
  prep(eW1, eW1t, 135, 64, 160, 64, 1);   // permuted edge input layout
  prep(eW2, eW2t, 64, 32, 64, 32, 0);
  prep(eW3, eW3t, 32, 16, 32, 16, 0);
  prep(cW1, cW1t, 16, 32, 32, 32, 0);
  prep(cW2, cW2t, 32, 32, 32, 32, 0);
  prep(cW3, cW3t, 32, 3, 32, 16, 0);
  prep(nW1, nW1t, 81, 128, 96, 128, 2);   // permuted node input layout
  prep(nW2, nW2t, 128, 128, 128, 128, 0);
  prep(nW3, nW3t, 128, 64, 128, 64, 0);

  embed_kernel<<<(N * 64 + 255) / 256, 256, 0, stream>>>(
      x, emb_W, emb_b, pos, H0, P0, (float*)d_out, N);

  hipMemsetAsync(cntf, 0, (size_t)N * 4, stream);
  count_kernel<<<(E + 255) / 256, 256, 0, stream>>>(eidx, cntf, E);

  float* Hin = H0; float* Hout = H1;
  float* Pin = P0; float* Pout = P1;
  const float dt = 2.0f / 3.0f;
  for (int s = 0; s < 3; ++s) {
    const float t0 = dt * (float)s;
    hipMemsetAsync(aggs, 0, (size_t)N * 16 * 4, stream);
    hipMemsetAsync(csum, 0, (size_t)N * 3 * 4, stream);
    edge_kernel<<<(E + 15) / 16, 32, 0, stream>>>(
        Hin, Pin, ea, eidx,
        eW1t, eb1, eW2t, eb2, eW3t, eb3,
        cW1t, cb1, cW2t, cb2, cW3t, cb3,
        aggs, csum, t0, E);
    node_kernel<<<(N + 15) / 16, 32, 0, stream>>>(
        Hin, Pin, aggs, csum, cntf,
        nW1t, nb1, nW2t, nb2, nW3t, nb3,
        Hout, Pout, (float*)d_out + (size_t)(s + 1) * N * 64, t0, dt, N);
    float* tmp = Hin; Hin = Hout; Hout = tmp;
    tmp = Pin; Pin = Pout; Pout = tmp;
  }
}
